// SingleImagePixelContrastLoss_18485539242910
// MI455X (gfx1250) — matrix operand order repaired and checked by reading, not compiled, not run on hardware
//
#include <hip/hip_runtime.h>

typedef __attribute__((ext_vector_type(16))) _Float16 v16h;
typedef __attribute__((ext_vector_type(8)))  _Float16 h8;
typedef __attribute__((ext_vector_type(4)))  _Float16 h4;
typedef __attribute__((ext_vector_type(8)))  float    v8f;
typedef __attribute__((ext_vector_type(4)))  float    f4;

#define IMG   4
#define A_    32
#define V_    128
#define D_    256
#define N_    (A_ * V_)      /* 4096 rows per image */
#define NBLK  (N_ / 16)      /* 256 column tiles per image */
#define INV_T (1.0f / 0.07f)

#define WAVES   4            /* waves per workgroup (64 rows/WG) */
#define COLPAD  264          /* 256 halves + 8 pad halves (16B) per column */
#define BUFH    (16 * COLPAD)
#define BUFB    (BUFH * 2)   /* bytes per LDS buffer = 8448 */

#ifndef USE_ASYNC_LDS
#define USE_ASYNC_LDS 1      /* CDNA5 global_load_async_to_lds_b128 staging */
#endif

/* ---------------- kernel 1: f32 -> f16 cast into workspace ---------------- */
__global__ void supcon_cast_f16(const float* __restrict__ src,
                                _Float16* __restrict__ dst, int n4) {
  int i = blockIdx.x * blockDim.x + threadIdx.x;
  if (i < n4) {
    f4 v = ((const f4*)src)[i];
    h4 h;
    h.x = (_Float16)v.x; h.y = (_Float16)v.y;
    h.z = (_Float16)v.z; h.w = (_Float16)v.w;
    ((h4*)dst)[i] = h;
  }
}

/* Issue this wave's share (4 columns, 512B each) of one column-tile stage. */
__device__ __forceinline__ void stage_issue(const _Float16* __restrict__ Xh,
                                            size_t imgOff, int j_base, int w,
                                            int lane, unsigned ldsByteOff,
                                            _Float16* ldsGeneric) {
#pragma unroll
  for (int i = 0; i < 4; ++i) {
    const int ct = w * 4 + i;
    const int c  = j_base + ct;                 /* col n -> (a=n%32, v=n/32) */
    const _Float16* gp =
        Xh + imgOff + ((size_t)(c & 31) * V_ + (c >> 5)) * D_ + lane * 8;
#if USE_ASYNC_LDS
    unsigned loff = ldsByteOff + (unsigned)(ct * (COLPAD * 2) + lane * 16);
    asm volatile("global_load_async_to_lds_b128 %0, %1, off"
                 :: "v"(loff), "v"(gp) : "memory");
    (void)ldsGeneric;
#else
    uint4 u = *(const uint4*)gp;
    *(uint4*)(ldsGeneric + ct * COLPAD + lane * 8) = u;
#endif
  }
}

__device__ __forceinline__ void stage_wait(bool more) {
#if USE_ASYNC_LDS
  if (more) asm volatile("s_wait_asynccnt 0x4" ::: "memory");
  else      asm volatile("s_wait_asynccnt 0x0" ::: "memory");
#else
  (void)more;
#endif
}

/* 16x16 logits tile from LDS columns: two independent WMMA chains. */
__device__ __forceinline__ v8f tile_gemm(const v16h* __restrict__ afrag,
                                         const _Float16* __restrict__ bp,
                                         int part) {
  v8f acc0 = {}, acc1 = {};
#pragma unroll
  for (int kc = 0; kc < 8; kc += 2) {
    v16h b0 = *(const v16h*)(bp + (kc + 0) * 32 + part * 16);
    acc0 = __builtin_amdgcn_wmma_f32_16x16x32_f16(
             false, afrag[kc + 0], false, b0, (short)0, acc0, false, false);
    v16h b1 = *(const v16h*)(bp + (kc + 1) * 32 + part * 16);
    acc1 = __builtin_amdgcn_wmma_f32_16x16x32_f16(
             false, afrag[kc + 1], false, b1, (short)0, acc1, false, false);
  }
  return acc0 + acc1;
}

/* -------- kernel 2: fused SupCon, 4 row-block waves share LDS tiles ------- */
__global__ __launch_bounds__(128) void supcon_rowblock(
    const _Float16* __restrict__ Xh,      /* [IMG][A][V][D] f16 */
    const int* __restrict__ labels,       /* [IMG][A] */
    float* __restrict__ partials) {       /* [IMG * N_/16] */
  __shared__ _Float16 ldsb[2][BUFH];
  __shared__ int slab[A_];

  const int tid  = threadIdx.x;
  const int w    = tid >> 5;
  const int lane = tid & 31;
  const int lm   = lane & 15;
  const int part = lane >> 4;

  const int blk      = blockIdx.x;
  const int b        = blk >> 6;          /* image */
  const int rowsBase = (blk & 63) * 64;   /* 64 rows per WG */
  const int m_base   = rowsBase + w * 16; /* this wave's 16-row block */

  if (tid < A_) slab[tid] = labels[b * A_ + tid];
  __syncthreads();

  const size_t imgOff = (size_t)b * A_ * V_ * D_;
  const unsigned lds0 = (unsigned)(size_t)(&ldsb[0][0]);

  /* ---- A fragments: this wave's 16 rows x 256 K, contiguous k = 16*part + s per chunk ---- */
  v16h afrag[8];
  {
    const int r = m_base + lm;
    const _Float16* rp = Xh + imgOff + ((size_t)(r & 31) * V_ + (r >> 5)) * D_;
#pragma unroll
    for (int kc = 0; kc < 8; ++kc) {
      h8 lo = *(const h8*)(rp + kc * 32 + part * 16);
      h8 hi = *(const h8*)(rp + kc * 32 + part * 16 + 8);
      afrag[kc] = __builtin_shufflevector(lo, hi,
                    0,1,2,3,4,5,6,7,8,9,10,11,12,13,14,15);
    }
  }

  /* ---- labels for this lane's 8 output rows + column labels ---- */
  const int rowBase = (m_base & 16) + part * 8;
  int rlab[8];
#pragma unroll
  for (int g = 0; g < 8; ++g) rlab[g] = slab[rowBase + g];
  const int clab0 = slab[lm];
  const int clab1 = slab[16 + lm];

  /* ---- anchor-match mask for pass-B compute skipping (per wave) ---- */
  bool match = false;
  {
    const int myLab = slab[lane];
    const int rb0 = (m_base & 16);
#pragma unroll
    for (int t = 0; t < 16; ++t) match |= (myLab == slab[rb0 + t]);
  }
  const unsigned mmask = (unsigned)__ballot((int)match);

  /* =================== PASS A: per-row negative sums =================== */
  float nsum[8];
#pragma unroll
  for (int g = 0; g < 8; ++g) nsum[g] = 0.0f;

  stage_issue(Xh, imgOff, 0, w, lane, lds0, &ldsb[0][0]);
  for (int jt = 0; jt < NBLK; ++jt) {
    const int cur  = jt & 1;
    const int nxt  = cur ^ 1;
    const bool more = (jt + 1 < NBLK);
    if (more)
      stage_issue(Xh, imgOff, (jt + 1) * 16, w, lane,
                  lds0 + (unsigned)(nxt * BUFB), &ldsb[nxt][0]);
    stage_wait(more);
    __syncthreads();                       /* tile jt fully in ldsb[cur] */

    v8f acc = tile_gemm(afrag, &ldsb[cur][lm * COLPAD], part);
    const int clab = (jt & 1) ? clab1 : clab0;
#pragma unroll
    for (int g = 0; g < 8; ++g) {
      float l = acc[g] * INV_T - INV_T;    /* shift by row max == 1/T */
      float e = __expf(l);
      nsum[g] += (rlab[g] != clab) ? e : 0.0f;
    }
    __syncthreads();                       /* done reading ldsb[cur] */
  }
#pragma unroll
  for (int g = 0; g < 8; ++g) {
    float v = nsum[g];
    v += __shfl_xor(v, 1, 32);
    v += __shfl_xor(v, 2, 32);
    v += __shfl_xor(v, 4, 32);
    v += __shfl_xor(v, 8, 32);
    nsum[g] = v;
  }

  /* ============ PASS B: positive log-prob sums =========== */
  float psum[8];
#pragma unroll
  for (int g = 0; g < 8; ++g) psum[g] = 0.0f;

  stage_issue(Xh, imgOff, 0, w, lane, lds0, &ldsb[0][0]);
  for (int jt = 0; jt < NBLK; ++jt) {
    const int cur  = jt & 1;
    const int nxt  = cur ^ 1;
    const bool more = (jt + 1 < NBLK);
    if (more)
      stage_issue(Xh, imgOff, (jt + 1) * 16, w, lane,
                  lds0 + (unsigned)(nxt * BUFB), &ldsb[nxt][0]);
    stage_wait(more);
    __syncthreads();

    const unsigned tileAnch = (jt & 1) ? 0xFFFF0000u : 0x0000FFFFu;
    if (mmask & tileAnch) {                /* per-wave compute skip */
      v8f acc = tile_gemm(afrag, &ldsb[cur][lm * COLPAD], part);
      const int clab = (jt & 1) ? clab1 : clab0;
      const int j_base = jt * 16;
#pragma unroll
      for (int g = 0; g < 8; ++g) {
        const int row = m_base + part * 8 + g;
        const bool pos = (rlab[g] == clab) && (row != (j_base + lm));
        float l = acc[g] * INV_T - INV_T;
        float t = l - __logf(__expf(l) + nsum[g]);
        psum[g] += pos ? t : 0.0f;
      }
    }
    __syncthreads();
  }
#pragma unroll
  for (int g = 0; g < 8; ++g) {
    float v = psum[g];
    v += __shfl_xor(v, 1, 32);
    v += __shfl_xor(v, 2, 32);
    v += __shfl_xor(v, 4, 32);
    v += __shfl_xor(v, 8, 32);
    psum[g] = v;
  }

  /* ---- per-row positive counts and wave partial ---- */
  float s = 0.0f;
  if (lm == 0) {
#pragma unroll
    for (int g = 0; g < 8; ++g) {
      int sc = 0;
      for (int t = 0; t < A_; ++t) sc += (slab[t] == rlab[g]) ? 1 : 0;
      s += psum[g] / (float)(V_ * sc - 1);
    }
  }
  s += __shfl_xor(s, 16, 32);
  if (lane == 0) partials[b * (N_ / 16) + (m_base >> 4)] = -s;
}

/* ---------------- kernel 3: deterministic final reduction ---------------- */
__global__ void supcon_reduce(const float* __restrict__ partials,
                              float* __restrict__ out) {
  __shared__ float sm[256];
  const int t = threadIdx.x;
  float s = 0.0f;
  for (int i = t; i < IMG * (N_ / 16); i += 256) s += partials[i];
  sm[t] = s;
  __syncthreads();
  for (int off = 128; off > 0; off >>= 1) {
    if (t < off) sm[t] += sm[t + off];
    __syncthreads();
  }
  if (t == 0) out[0] = sm[0] / (float)(N_ * IMG);
}

extern "C" void kernel_launch(void* const* d_in, const int* in_sizes, int n_in,
                              void* d_out, int out_size, void* d_ws, size_t ws_size,
                              hipStream_t stream) {
  const float* feats  = (const float*)d_in[0];   /* [4,32,128,256] f32 */
  const int*   labels = (const int*)d_in[1];     /* [4,32] i32 */
  float*       out    = (float*)d_out;

  _Float16* Xh       = (_Float16*)d_ws;
  float*    partials = (float*)((char*)d_ws +
                        (size_t)IMG * A_ * V_ * D_ * sizeof(_Float16));

  const int nElem = IMG * A_ * V_ * D_;          /* 4,194,304 */
  const int n4    = nElem / 4;

  supcon_cast_f16<<<(n4 + 255) / 256, 256, 0, stream>>>(feats, Xh, n4);
  supcon_rowblock<<<IMG * (N_ / 64), 128, 0, stream>>>(Xh, labels, partials);
  supcon_reduce<<<1, 256, 0, stream>>>(partials, out);
}
